// BiA_Attention_88751204205348
// MI455X (gfx1250) — matrix operand order repaired and checked by reading, not compiled, not run on hardware
//
#include <hip/hip_runtime.h>
#include <hip/hip_bf16.h>
#include <math.h>

typedef __bf16 bf16;
typedef __attribute__((ext_vector_type(16))) __bf16 v16bf;
typedef __attribute__((ext_vector_type(8)))  __bf16 bf16x8;
typedef __attribute__((ext_vector_type(8)))  float  v8f;
typedef __attribute__((ext_vector_type(4)))  int    v4i_;

#define B_     4
#define C_     512
#define H_     56
#define W_     56
#define HEADS_ 8
#define HC_    64
#define GC_    256
#define NS_    784
#define NSP_   800
#define NW_    49
#define WH_    7
#define HW_    3136
#define NPIX_  12544        // B*H*W
#define NPOS_  153664       // B*NW*NS
#define WNS_   38416        // NW*NS
#define SCALE_ 0.044194173824159216f   // 512^-0.5
#define BN_INV_ 0.9999950000374997f    // 1/sqrt(1+1e-5)

// ---- WMMA fragment helpers (wave32, 16x16x32 bf16) -------------------------
// A (16xK tile, row-major, ld elements): lane m = l&15; two 8-elem K runs at
// (l>>4)*8 and 16+(l>>4)*8 (ISA 7.12.2, 16-bit A 16x32 table).
static __device__ __forceinline__ v16bf load_fragA(const bf16* tile, int ld, int lane) {
  const bf16* r = tile + (size_t)(lane & 15) * ld + ((lane >> 4) << 3);
  bf16x8 lo = *(const bf16x8*)(r);
  bf16x8 hi = *(const bf16x8*)(r + 16);
  v16bf f;
#pragma unroll
  for (int i = 0; i < 8; ++i) { f[i] = lo[i]; f[i + 8] = hi[i]; }
  return f;
}
// B (Kx16 tile) supplied as BT row-major [n][k]: lane n = l&15; two 8-elem K
// runs at (l>>4)*8 and 16+(l>>4)*8 (same K order as fragA).
static __device__ __forceinline__ v16bf load_fragB(const bf16* tileT, int ld, int lane) {
  const bf16* r = tileT + (size_t)(lane & 15) * ld + ((lane >> 4) << 3);
  bf16x8 lo = *(const bf16x8*)(r);
  bf16x8 hi = *(const bf16x8*)(r + 16);
  v16bf f;
#pragma unroll
  for (int i = 0; i < 8; ++i) { f[i] = lo[i]; f[i + 8] = hi[i]; }
  return f;
}
static __device__ __forceinline__ v8f wmma_bf16(v16bf a, v16bf b, v8f c) {
  return __builtin_amdgcn_wmma_f32_16x16x32_bf16(false, a, false, b, (short)0, c, false, false);
}
static __device__ __forceinline__ v8f vzero8() {
  v8f z = {0.f, 0.f, 0.f, 0.f, 0.f, 0.f, 0.f, 0.f};
  return z;
}

// ---- gfx1250 async global->LDS staging (with portable fallback) ------------
#if defined(__gfx1250__) && __has_builtin(__builtin_amdgcn_global_load_async_to_lds_b128)
#define HAVE_ASYNC_LDS 1
typedef __attribute__((address_space(1))) v4i_* as1_v4i;
typedef __attribute__((address_space(3))) v4i_* as3_v4i;
#else
#define HAVE_ASYNC_LDS 0
#endif

static __device__ __forceinline__ void async_wait0() {
#if HAVE_ASYNC_LDS
#if __has_builtin(__builtin_amdgcn_s_wait_asynccnt)
  __builtin_amdgcn_s_wait_asynccnt(0);
#else
  asm volatile("s_wait_asynccnt 0x0" ::: "memory");
#endif
#endif
}

// copy `n16` 16-byte chunks global->LDS cooperatively (blockDim.x threads)
static __device__ __forceinline__ void stage_to_lds(bf16* lds, const bf16* gsrc, int n16) {
#if HAVE_ASYNC_LDS
  for (int i = threadIdx.x; i < n16; i += blockDim.x) {
    v4i_* g = (v4i_*)(gsrc + i * 8);   // const-cast + retype (generic)
    v4i_* l = (v4i_*)(lds + i * 8);
    __builtin_amdgcn_global_load_async_to_lds_b128((as1_v4i)g, (as3_v4i)l, 0, 0);
  }
  async_wait0();
#else
  for (int i = threadIdx.x; i < n16; i += blockDim.x) {
    *(bf16x8*)(lds + i * 8) = *(const bf16x8*)(gsrc + i * 8);
  }
#endif
}

// ---- small elementwise kernels ---------------------------------------------
__global__ void k_f2b(const float* __restrict__ src, bf16* __restrict__ dst, long n) {
  long i = (long)blockIdx.x * blockDim.x + threadIdx.x;
  if (i < n) dst[i] = (bf16)src[i];
}

// x (B,C,H,W) fp32 -> pixel-major bf16 [B*H*W, C] via LDS 32x32 tile transpose
__global__ __launch_bounds__(256) void k_xT(const float* __restrict__ x,
                                            bf16* __restrict__ xT) {
  __shared__ float tile[32][33];
  int p0 = blockIdx.x * 32;          // pixel tile (HW_/32 = 98)
  int c0 = blockIdx.y * 32;          // channel tile (C_/32 = 16)
  int b  = blockIdx.z;
  int tx = threadIdx.x;              // 0..31
#pragma unroll
  for (int r = threadIdx.y; r < 32; r += 8) {
    tile[r][tx] = x[((size_t)b * C_ + c0 + r) * HW_ + p0 + tx];  // coalesced in p
  }
  __syncthreads();
#pragma unroll
  for (int r = threadIdx.y; r < 32; r += 8) {
    xT[((size_t)b * HW_ + p0 + r) * C_ + c0 + tx] = (bf16)tile[tx][r];  // coalesced in c
  }
}

// depthwise 3x3 stride-2 pad-1 (optionally fused BN(eval)+exact GELU)
__global__ void k_dwconv(const float* __restrict__ in, const float* __restrict__ wt,
                         float* __restrict__ out, int Hi, int Wi, int Ho, int Wo,
                         int do_act, const float* __restrict__ gamma,
                         const float* __restrict__ beta) {
  long id = (long)blockIdx.x * blockDim.x + threadIdx.x;
  long total = 8L * GC_ * Ho * Wo;
  if (id >= total) return;
  int xo = (int)(id % Wo); long t = id / Wo;
  int yo = (int)(t % Ho);  t /= Ho;
  int c  = (int)(t % GC_); int bg = (int)(t / GC_);
  const float* src = in + ((long)bg * GC_ + c) * Hi * Wi;
  const float* wp  = wt + c * 9;
  float acc = 0.f;
#pragma unroll
  for (int ky = 0; ky < 3; ++ky) {
    int iy = 2 * yo - 1 + ky;
    if (iy < 0 || iy >= Hi) continue;
#pragma unroll
    for (int kx = 0; kx < 3; ++kx) {
      int ix = 2 * xo - 1 + kx;
      if (ix < 0 || ix >= Wi) continue;
      acc += src[iy * Wi + ix] * wp[ky * 3 + kx];
    }
  }
  if (do_act) {
    acc = acc * BN_INV_ * gamma[c] + beta[c];
    acc = 0.5f * acc * (1.0f + erff(acc * 0.70710678118654752f));
  }
  out[id] = acc;
}

// 1x1 offset projection + tanh*range*FACTOR + grid -> pixel-space sample coords
__global__ void k_offsets(const float* __restrict__ o3, const float* __restrict__ off_w,
                          float* __restrict__ gy, float* __restrict__ gx) {
  long id = (long)blockIdx.x * blockDim.x + threadIdx.x;
  if (id >= 8L * NW_ * NS_ * 2) return;
  int dim = (int)(id & 1); long t = id >> 1;
  int s  = (int)(t % NS_); t /= NS_;
  int nw = (int)(t % NW_); int bg = (int)(t / NW_);
  const float* ov = o3 + (long)bg * GC_ * NW_ + nw;          // channel stride NW_
  const float* wv = off_w + ((long)dim * NS_ + s) * GC_;
  float acc = 0.f;
  for (int c = 0; c < GC_; ++c) acc += ov[(long)c * NW_] * wv[c];
  float off = tanhf(acc) * (2.0f / 56.0f);                   // tanh * (1/H) * FACTOR
  int idx = (dim == 0) ? (s / 28) : (s % 28);
  float gridv = (2.0f * (float)idx) / 55.0f * 2.0f - 1.0f;   // align_corners grid
  float pix = (off + gridv + 1.0f) * 0.5f * 55.0f;
  long o = ((long)bg * NW_ + nw) * NS_ + s;
  if (dim == 0) gy[o] = pix; else gx[o] = pix;
}

// bilinear grid_sample (zero padding) -> bf16 samp [pos][C]
__global__ __launch_bounds__(256) void k_sample(const float* __restrict__ x,
                                                const float* __restrict__ gy,
                                                const float* __restrict__ gx,
                                                bf16* __restrict__ samp) {
  long pos = blockIdx.x;                        // (bg, nw, s) flat
  int s = (int)(pos % NS_); long t = pos / NS_;
  int nw = (int)(t % NW_); int bg = (int)(t / NW_);
  int b = bg >> 1, g = bg & 1;
  float fy = gy[pos], fx = gx[pos];
  float y0f = floorf(fy), x0f = floorf(fx);
  float wy = fy - y0f, wx = fx - x0f;
  int y0 = (int)y0f, x0 = (int)x0f;
  float wgt[4] = {(1.f - wy) * (1.f - wx), (1.f - wy) * wx, wy * (1.f - wx), wy * wx};
  int ys[4] = {y0, y0, y0 + 1, y0 + 1};
  int xs[4] = {x0, x0 + 1, x0, x0 + 1};
  int c = threadIdx.x;
  const float* src = x + ((long)b * C_ + g * GC_ + c) * HW_;
  float acc = 0.f;
#pragma unroll
  for (int tap = 0; tap < 4; ++tap) {
    int yy = ys[tap], xx = xs[tap];
    if (yy < 0 || yy >= H_ || xx < 0 || xx >= W_) continue;
    acc += src[yy * W_ + xx] * wgt[tap];
  }
  samp[(((long)b * NW_ + nw) * NS_ + s) * C_ + g * GC_ + c] = (bf16)acc;
}

__global__ void k_zpad(bf16* __restrict__ vb) {
  long i = (long)blockIdx.x * blockDim.x + threadIdx.x;
  if (i >= 100352L * 16) return;                // 32*49*64 rows x 16 pad cols
  long row = i >> 4; int cc = (int)(i & 15);
  vb[row * NSP_ + NS_ + cc] = (bf16)0.0f;
}

// ---- generic 512-K WMMA GEMM, B slab staged in LDS via async copy ----------
// A [M,512] bf16 row-major, BT [512,512] bf16 (out-channel-major), bias fp32.
// mode 0: Q  -> outf = q fp32 NCHW;  outb = wq bf16 window-ordered, *SCALE
// mode 1: K  -> outb = kb  [(bh*NW+nw)*NS+s][hc]
// mode 2: V  -> outb = vb  [(bh*NW+nw)*64+hc][NSP] (transposed, padded)
// mode 3: P  -> outf = d_out fp32 NCHW
__global__ __launch_bounds__(256) void k_gemm512(const bf16* __restrict__ A,
                                                 const bf16* __restrict__ BT,
                                                 const float* __restrict__ bias,
                                                 float* __restrict__ outf,
                                                 bf16* __restrict__ outb,
                                                 long M, int mode) {
  __shared__ bf16 Bs[64 * C_];                   // 64 KB: BT rows n0..n0+63
  int lane = threadIdx.x & 31;
  int wave = threadIdx.x >> 5;
  int n0 = blockIdx.y * 64;
  // stage the 64x512 weight slab once per block (async global->LDS)
  stage_to_lds(Bs, BT + (size_t)n0 * C_, 64 * C_ / 8);
  __syncthreads();
  long m0 = (long)blockIdx.x * 128 + wave * 16;
  if (m0 + 16 > M) return;
  v8f acc[4];
#pragma unroll
  for (int j = 0; j < 4; ++j) acc[j] = vzero8();
  const bf16* Ap = A + m0 * C_;
  for (int k0 = 0; k0 < C_; k0 += 32) {
    v16bf a = load_fragA(Ap + k0, C_, lane);
#pragma unroll
    for (int j = 0; j < 4; ++j) {
      v16bf b = load_fragB(Bs + (size_t)(16 * j) * C_ + k0, C_, lane);
      acc[j] = wmma_bf16(a, b, acc[j]);
    }
  }
#pragma unroll
  for (int j = 0; j < 4; ++j) {
    int n = n0 + 16 * j + (lane & 15);
    float bv = bias[n];
#pragma unroll
    for (int r = 0; r < 8; ++r) {
      long m = m0 + r + ((lane >> 4) << 3);
      float val = acc[j][r] + bv;
      if (mode == 0) {
        long b = m / HW_; int p = (int)(m - b * HW_);
        outf[(b * C_ + n) * HW_ + p] = val;
        int head = n >> 6, hc = n & 63;
        int row = p / W_, col = p - row * W_;
        int w = (row >> 3) * WH_ + (col >> 3);
        int qq = ((row & 7) << 3) + (col & 7);
        outb[(((b * HEADS_ + head) * NW_ + w) * 64L + qq) * HC_ + hc] =
            (bf16)(val * SCALE_);
      } else if (mode == 1) {
        long b = m / WNS_; long rem = m - b * WNS_;
        int head = n >> 6, hc = n & 63;
        outb[((b * HEADS_ + head) * (long)WNS_ + rem) * HC_ + hc] = (bf16)val;
      } else if (mode == 2) {
        long b = m / WNS_; long rem = m - b * WNS_;
        int nw = (int)(rem / NS_); int s = (int)(rem - (long)nw * NS_);
        int head = n >> 6, hc = n & 63;
        outb[(((b * HEADS_ + head) * NW_ + nw) * 64L + hc) * NSP_ + s] = (bf16)val;
      } else {
        long b = m / HW_; int p = (int)(m - b * HW_);
        outf[(b * C_ + n) * HW_ + p] = val;
      }
    }
  }
}

// ---- fused attention: scores(WMMA)+bias -> softmax(LDS) -> P*V^T(WMMA) -----
// one block per (bh = b*8+head, window w); 128 threads = 4 waves.
// phase 1: s-tiles partitioned across waves (kb read once per block)
// phase 3: hc-tiles partitioned across waves (vb read once per block)
__global__ __launch_bounds__(128) void k_attention(const bf16* __restrict__ wqb,
                                                   const bf16* __restrict__ kb,
                                                   const bf16* __restrict__ vb,
                                                   const float* __restrict__ posembed,
                                                   bf16* __restrict__ aout) {
  extern __shared__ char smem[];
  float* attn = (float*)smem;                                  // [64][NS_]
  bf16* pb = (bf16*)(smem + (size_t)64 * NS_ * sizeof(float)); // [64][NSP_]
  int bh = blockIdx.x / NW_;
  int w  = blockIdx.x - bh * NW_;
  int head = bh & 7;
  int lane = threadIdx.x & 31;
  int wave = threadIdx.x >> 5;
  int wr = w / WH_, wc = w - wr * WH_;
  const bf16* qbase = wqb + (size_t)blockIdx.x * 64 * HC_;
  const bf16* kbase = kb + (size_t)blockIdx.x * NS_ * HC_;
  const bf16* vbase = vb + (size_t)blockIdx.x * 64 * NSP_;

  // phase 1: all 8 wq A-fragments per wave; disjoint s-tiles per wave
  v16bf aq[4][2];
#pragma unroll
  for (int mt = 0; mt < 4; ++mt) {
    aq[mt][0] = load_fragA(qbase + (size_t)(mt * 16) * HC_, HC_, lane);
    aq[mt][1] = load_fragA(qbase + (size_t)(mt * 16) * HC_ + 32, HC_, lane);
  }
  for (int st = wave; st < NS_ / 16; st += 4) {
    int s0 = st * 16;
    v16bf b0 = load_fragB(kbase + (size_t)s0 * HC_, HC_, lane);
    v16bf b1 = load_fragB(kbase + (size_t)s0 * HC_ + 32, HC_, lane);
    int s = s0 + (lane & 15);
    int kr = (s / 28) * 2, kc = (s - (s / 28) * 28) * 2;
#pragma unroll
    for (int mt = 0; mt < 4; ++mt) {
      v8f acc = vzero8();
      acc = wmma_bf16(aq[mt][0], b0, acc);
      acc = wmma_bf16(aq[mt][1], b1, acc);
#pragma unroll
      for (int r = 0; r < 8; ++r) {
        int qq = mt * 16 + r + ((lane >> 4) << 3);
        int qr = wr * 8 + (qq >> 3), qc = wc * 8 + (qq & 7);
        float bias = posembed[(size_t)head * 111 * 111 +
                              (size_t)(kr - qr + 55) * 111 + (kc - qc + 55)];
        attn[qq * NS_ + s] = acc[r] + bias;
      }
    }
  }
  // zero K-pad columns of probability buffer (NS_..NSP_)
  for (int i = threadIdx.x; i < 64 * (NSP_ - NS_); i += 128) {
    int row = i >> 4, cc = i & 15;
    pb[row * NSP_ + NS_ + cc] = (bf16)0.0f;
  }
  __syncthreads();
  // phase 2: softmax, one thread per q row
  if (threadIdx.x < 64) {
    int row = threadIdx.x;
    const float* arow = attn + row * NS_;
    float mx = -3.4e38f;
    for (int s = 0; s < NS_; ++s) mx = fmaxf(mx, arow[s]);
    float sum = 0.f;
    for (int s = 0; s < NS_; ++s) sum += expf(arow[s] - mx);
    float inv = 1.0f / sum;
    bf16* prow = pb + row * NSP_;
    for (int s = 0; s < NS_; ++s) prow[s] = (bf16)(expf(arow[s] - mx) * inv);
  }
  __syncthreads();
  // phase 3: out = P [64 x NSP] x V^T [NSP x 64]; wave owns hc-tile = wave
  v8f acc2[4];
#pragma unroll
  for (int mt = 0; mt < 4; ++mt) acc2[mt] = vzero8();
  const bf16* vcol = vbase + (size_t)(wave * 16) * NSP_;
  for (int k0 = 0; k0 < NSP_; k0 += 32) {
    v16bf b = load_fragB(vcol + k0, NSP_, lane);
#pragma unroll
    for (int mt = 0; mt < 4; ++mt) {
      v16bf a = load_fragA(pb + (size_t)(mt * 16) * NSP_ + k0, NSP_, lane);
      acc2[mt] = wmma_bf16(a, b, acc2[mt]);
    }
  }
  long b = bh >> 3;
  int hc = wave * 16 + (lane & 15);
#pragma unroll
  for (int mt = 0; mt < 4; ++mt) {
#pragma unroll
    for (int r = 0; r < 8; ++r) {
      int qq = mt * 16 + r + ((lane >> 4) << 3);
      int qr = wr * 8 + (qq >> 3), qc = wc * 8 + (qq & 7);
      long pix = b * HW_ + (long)qr * W_ + qc;
      aout[pix * C_ + head * HC_ + hc] = (bf16)acc2[mt][r];
    }
  }
}

// ---- host launcher ----------------------------------------------------------
extern "C" void kernel_launch(void* const* d_in, const int* in_sizes, int n_in,
                              void* d_out, int out_size, void* d_ws, size_t ws_size,
                              hipStream_t stream) {
  (void)in_sizes; (void)n_in; (void)out_size; (void)ws_size;
  const float* x     = (const float*)d_in[0];
  const float* Wq    = (const float*)d_in[1];
  const float* bq    = (const float*)d_in[2];
  const float* Wk    = (const float*)d_in[3];
  const float* bk    = (const float*)d_in[4];
  const float* Wv    = (const float*)d_in[5];
  const float* bv    = (const float*)d_in[6];
  const float* dw1   = (const float*)d_in[7];
  const float* dw2   = (const float*)d_in[8];
  const float* dw3   = (const float*)d_in[9];
  const float* gamma = (const float*)d_in[10];
  const float* beta  = (const float*)d_in[11];
  const float* off_w = (const float*)d_in[12];
  const float* pose  = (const float*)d_in[13];
  const float* Wp    = (const float*)d_in[14];
  const float* bp    = (const float*)d_in[15];
  float* out = (float*)d_out;

  char* ws = (char*)d_ws;
  size_t off = 0;
  auto alloc = [&](size_t bytes) -> void* {
    void* p = ws + off;
    off = (off + bytes + 255) & ~(size_t)255;
    return p;
  };
  bf16*  xT   = (bf16*)alloc((size_t)NPIX_ * C_ * 2);
  bf16*  Wqb  = (bf16*)alloc((size_t)C_ * C_ * 2);
  bf16*  Wkb  = (bf16*)alloc((size_t)C_ * C_ * 2);
  bf16*  Wvb  = (bf16*)alloc((size_t)C_ * C_ * 2);
  bf16*  Wpb  = (bf16*)alloc((size_t)C_ * C_ * 2);
  float* qf   = (float*)alloc((size_t)B_ * C_ * HW_ * 4);
  bf16*  wqb  = (bf16*)alloc((size_t)32 * NW_ * 64 * HC_ * 2);
  float* o1   = (float*)alloc(8L * GC_ * 28 * 28 * 4);
  float* o2   = (float*)alloc(8L * GC_ * 14 * 14 * 4);
  float* o3   = (float*)alloc(8L * GC_ * 7 * 7 * 4);
  float* gy   = (float*)alloc(8L * NW_ * NS_ * 4);
  float* gx   = (float*)alloc(8L * NW_ * NS_ * 4);
  bf16*  samp = (bf16*)alloc((size_t)NPOS_ * C_ * 2);
  bf16*  kb   = (bf16*)alloc((size_t)32 * NW_ * NS_ * HC_ * 2);
  bf16*  vb   = (bf16*)alloc((size_t)32 * NW_ * 64 * NSP_ * 2);
  bf16*  aout = (bf16*)alloc((size_t)NPIX_ * C_ * 2);

  auto cdiv = [](long a, long b) { return (unsigned)((a + b - 1) / b); };

  k_f2b<<<cdiv(262144, 256), 256, 0, stream>>>(Wq, Wqb, 262144);
  k_f2b<<<cdiv(262144, 256), 256, 0, stream>>>(Wk, Wkb, 262144);
  k_f2b<<<cdiv(262144, 256), 256, 0, stream>>>(Wv, Wvb, 262144);
  k_f2b<<<cdiv(262144, 256), 256, 0, stream>>>(Wp, Wpb, 262144);

  dim3 gt(HW_ / 32, C_ / 32, B_);
  k_xT<<<gt, dim3(32, 8), 0, stream>>>(x, xT);

  dim3 gq(cdiv(NPIX_, 128), 8);
  k_gemm512<<<gq, 256, 0, stream>>>(xT, Wqb, bq, qf, wqb, NPIX_, 0);

  k_dwconv<<<cdiv(8L * GC_ * 28 * 28, 256), 256, 0, stream>>>(qf, dw1, o1, 56, 56, 28, 28, 0, nullptr, nullptr);
  k_dwconv<<<cdiv(8L * GC_ * 14 * 14, 256), 256, 0, stream>>>(o1, dw2, o2, 28, 28, 14, 14, 0, nullptr, nullptr);
  k_dwconv<<<cdiv(8L * GC_ * 7 * 7, 256), 256, 0, stream>>>(o2, dw3, o3, 14, 14, 7, 7, 1, gamma, beta);
  k_offsets<<<cdiv(8L * NW_ * NS_ * 2, 256), 256, 0, stream>>>(o3, off_w, gy, gx);
  k_sample<<<8 * NW_ * NS_, 256, 0, stream>>>(x, gy, gx, samp);

  dim3 gkv(cdiv(NPOS_, 128), 8);
  k_gemm512<<<gkv, 256, 0, stream>>>(samp, Wkb, bk, nullptr, kb, NPOS_, 1);
  k_gemm512<<<gkv, 256, 0, stream>>>(samp, Wvb, bv, nullptr, vb, NPOS_, 2);
  k_zpad<<<cdiv(100352L * 16, 256), 256, 0, stream>>>(vb);

  size_t smem = (size_t)64 * NS_ * 4 + (size_t)64 * NSP_ * 2;  // 303,104 B
  k_attention<<<32 * NW_, 128, smem, stream>>>(wqb, kb, vb, pose, aout);

  k_gemm512<<<gq, 256, 0, stream>>>(aout, Wpb, bp, out, nullptr, NPIX_, 3);
}
